// MultiLevelCompressionSynthesizer_77833397338711
// MI455X (gfx1250) — matrix operand order repaired and checked by reading, not compiled, not run on hardware
//
#include <hip/hip_runtime.h>
#include <hip/hip_bf16.h>

// ---------------------------------------------------------------------------
// CDNA5 (gfx1250) bf16 WMMA pipeline for hierarchical VQ autoencoder forward.
// Staging copies use the gfx1250 async global->LDS engine (ASYNCcnt).
// ---------------------------------------------------------------------------

typedef __bf16 bf16_t;
typedef bf16_t v16bf __attribute__((ext_vector_type(16)));
typedef float  v8f   __attribute__((ext_vector_type(8)));

union FragB16 { v16bf v; uint4 q[2]; };

__device__ __forceinline__ unsigned short f2bf(float f) {
  unsigned int u = __float_as_uint(f);
  u += 0x7FFFu + ((u >> 16) & 1u);   // round to nearest even
  return (unsigned short)(u >> 16);
}

// gfx1250 async copy: LDS[lds_off] = MEM[gaddr] (8 bytes per lane), ASYNCcnt.
__device__ __forceinline__ void async_ld_b64(unsigned lds_off, const void* gaddr) {
  asm volatile("global_load_async_to_lds_b64 %0, %1, off"
               :: "v"(lds_off), "v"(gaddr) : "memory");
}
__device__ __forceinline__ void wait_async0() {
  asm volatile("s_wait_asynccnt 0x0" ::: "memory");
}

// ---------------- conversion / prep kernels ----------------

__global__ void k_cvt_bf16(const float* __restrict__ in,
                           unsigned short* __restrict__ out, int n) {
  int i = blockIdx.x * blockDim.x + threadIdx.x;
  if (i < n) out[i] = f2bf(in[i]);
}

// W[k][m] (f32, row-major KxM) -> Wt[m][k] (bf16, row-major MxK)
__global__ void k_transpose_bf16(const float* __restrict__ W,
                                 unsigned short* __restrict__ Wt, int K, int M) {
  int i = blockIdx.x * blockDim.x + threadIdx.x;
  if (i < K * M) {
    int k = i / M, m = i - k * M;
    Wt[(size_t)m * K + k] = f2bf(W[i]);
  }
}

// per-code squared norm, one wave per codebook row
__global__ __launch_bounds__(256)
void k_norms(const float* __restrict__ CB, float* __restrict__ nrm, int d) {
  const int wave = (blockIdx.x * blockDim.x + threadIdx.x) >> 5;
  const int lane = threadIdx.x & 31;
  const float* row = CB + (size_t)wave * d;
  float s = 0.f;
  for (int c = lane; c < d; c += 32) { float v = row[c]; s += v * v; }
  for (int off = 16; off; off >>= 1) s += __shfl_xor(s, off, 32);
  if (lane == 0) nrm[wave] = s;
}

__global__ void k_init_loss(float* acc) { if (threadIdx.x < 4) acc[threadIdx.x] = 0.f; }

// ---------------- WMMA GEMM: out = act(A[N,K] * Wt[M,K]^T + bias) ----------
// block = 256 thr = 8 waves; block computes rows [16*bx,16*bx+16) x cols
// [128*by, 128*by+128). A chunk (16x64 bf16) staged in LDS via async copy,
// shared by all 8 waves.

template<int RELU, int WF32, int WB16>
__global__ __launch_bounds__(256)
void k_gemm(const unsigned short* __restrict__ A,
            const unsigned short* __restrict__ Wt,
            const float* __restrict__ bias,
            float* __restrict__ outF, unsigned short* __restrict__ outB,
            int K, int M) {
  __shared__ __align__(16) unsigned short As[16 * 64];
  const int tid   = threadIdx.x;
  const int wtile = tid >> 5;
  const int lane  = tid & 31;
  const int hi    = lane >> 4;
  const int ln    = lane & 15;
  const int rowBase = blockIdx.x * 16;
  const int n0 = blockIdx.y * 128 + wtile * 16 + ln;
  const unsigned short* wrow = Wt + (size_t)n0 * K;
  const int khA = hi * 8;    // A: K split {0..7,16..23} / {8..15,24..31}
  const int khB = hi * 8;    // B: K split {0..7,16..23} / {8..15,24..31}
  const unsigned asOff = (unsigned)(size_t)As + tid * 8u;     // LDS byte offset
  const unsigned short* aSrc =
      A + (size_t)(rowBase + (tid >> 4)) * K + ((tid & 15) << 2);
  v8f acc = {};
  for (int k0 = 0; k0 < K; k0 += 64) {
    __syncthreads();
    // async stage A chunk: 16 rows x 64 k of bf16; one b64 per thread
    async_ld_b64(asOff, aSrc + k0);
    wait_async0();
    __syncthreads();
    __builtin_prefetch(wrow + k0 + 64, 0, 0);
#pragma unroll
    for (int ks = 0; ks < 64; ks += 32) {
      FragB16 a, b;
      const unsigned short* ap = As + ln * 64 + ks + khA;
      a.q[0] = *(const uint4*)(ap);
      a.q[1] = *(const uint4*)(ap + 16);
      const unsigned short* bp = wrow + k0 + ks + khB;
      b.q[0] = *(const uint4*)(bp);
      b.q[1] = *(const uint4*)(bp + 16);
      acc = __builtin_amdgcn_wmma_f32_16x16x32_bf16(false, a.v, false, b.v,
                                                    (short)0, acc, false, false);
    }
  }
  const float bn = bias[n0];
#pragma unroll
  for (int v = 0; v < 8; ++v) {
    float val = acc[v] + bn;
    if (RELU) val = fmaxf(val, 0.f);
    const size_t o = (size_t)(rowBase + v + hi * 8) * M + n0;
    if (WF32) outF[o] = val;
    if (WB16) outB[o] = f2bf(val);
  }
}

// ---------------- LayerNorm (wave per row), writes f32 + bf16 --------------

__global__ __launch_bounds__(256)
void k_layernorm(const float* __restrict__ X, const float* __restrict__ g,
                 const float* __restrict__ be, float* __restrict__ outF,
                 unsigned short* __restrict__ outB, int d) {
  const int wave = (blockIdx.x * blockDim.x + threadIdx.x) >> 5;
  const int lane = threadIdx.x & 31;
  const float* row = X + (size_t)wave * d;
  float xs[16];
  const int nit = d >> 5;          // d in {128,256,512} -> 4..16
  float s = 0.f, ss = 0.f;
  for (int i = 0; i < nit; ++i) {
    float v = row[lane + (i << 5)];
    xs[i] = v; s += v; ss += v * v;
  }
  for (int off = 16; off; off >>= 1) {
    s  += __shfl_xor(s, off, 32);
    ss += __shfl_xor(ss, off, 32);
  }
  const float inv = 1.f / (float)d;
  const float mu  = s * inv;
  const float var = ss * inv - mu * mu;
  const float rs  = rsqrtf(var + 1e-5f);
  for (int i = 0; i < nit; ++i) {
    int c = lane + (i << 5);
    float val = (xs[i] - mu) * rs * g[c] + be[c];
    outF[(size_t)wave * d + c] = val;
    outB[(size_t)wave * d + c] = f2bf(val);
  }
}

// ---------------- VQ argmin via WMMA --------------------------------------
// block = 16 rows x 8192 codes. z-tile async-staged in LDS; each wave sweeps
// 64 code tiles; score = 2*z.e - |e|^2 maximized (== distance minimized).

__global__ __launch_bounds__(256)
void k_vq_argmin(const unsigned short* __restrict__ Zb,
                 const unsigned short* __restrict__ CBb,
                 const float* __restrict__ nrm,
                 int* __restrict__ idxOut, int d) {
  __shared__ __align__(16) unsigned short Zs[16 * 512];
  __shared__ float redV[8][16];
  __shared__ int   redI[8][16];
  const int tid  = threadIdx.x;
  const int wave = tid >> 5;
  const int lane = tid & 31;
  const int hi   = lane >> 4;
  const int ln   = lane & 15;
  const int rowBase = blockIdx.x * 16;
  {  // async stage z tile: 16*d bf16, contiguous span, 8B chunks
    const unsigned zsBase = (unsigned)(size_t)Zs;
    const char* src = (const char*)(Zb + (size_t)rowBase * d);
    const int tot8 = (16 * d) >> 2;   // (16*d*2)/8 chunks
    for (int i = tid; i < tot8; i += 256)
      async_ld_b64(zsBase + (unsigned)i * 8u, src + (size_t)i * 8u);
    wait_async0();
  }
  __syncthreads();
  float bv[8];
  int   bi[8];
#pragma unroll
  for (int v = 0; v < 8; ++v) { bv[v] = -3.4e38f; bi[v] = 0; }
  const int khA = hi * 8, khB = hi * 8;
  for (int tile = wave; tile < 512; tile += 8) {
    const int n0 = tile * 16 + ln;
    const unsigned short* crow = CBb + (size_t)n0 * d;
    v8f acc = {};
    for (int k = 0; k < d; k += 32) {
      FragB16 a, b;
      const unsigned short* ap = Zs + ln * d + k + khA;
      a.q[0] = *(const uint4*)(ap);
      a.q[1] = *(const uint4*)(ap + 16);
      const unsigned short* bp = crow + k + khB;
      b.q[0] = *(const uint4*)(bp);
      b.q[1] = *(const uint4*)(bp + 16);
      acc = __builtin_amdgcn_wmma_f32_16x16x32_bf16(false, a.v, false, b.v,
                                                    (short)0, acc, false, false);
    }
    const float nn = nrm[n0];
#pragma unroll
    for (int v = 0; v < 8; ++v) {
      float s = 2.f * acc[v] - nn;
      if (s > bv[v]) { bv[v] = s; bi[v] = n0; }
    }
  }
  // reduce across the 16-lane half-groups (each group covers N=0..15)
#pragma unroll
  for (int v = 0; v < 8; ++v) {
    for (int off = 8; off; off >>= 1) {
      float ov = __shfl_xor(bv[v], off, 16);
      int   oi = __shfl_xor(bi[v], off, 16);
      if (ov > bv[v] || (ov == bv[v] && oi < bi[v])) { bv[v] = ov; bi[v] = oi; }
    }
  }
  if (ln == 0) {
#pragma unroll
    for (int v = 0; v < 8; ++v) {
      redV[wave][v + hi * 8] = bv[v];
      redI[wave][v + hi * 8] = bi[v];
    }
  }
  __syncthreads();
  if (tid < 16) {
    float best = redV[0][tid];
    int  bidx = redI[0][tid];
    for (int w = 1; w < 8; ++w) {
      float ov = redV[w][tid]; int oi = redI[w][tid];
      if (ov > best || (ov == best && oi < bidx)) { best = ov; bidx = oi; }
    }
    idxOut[rowBase + tid] = bidx;
  }
}

// ---------------- gather q + commitment loss (wave per row) ----------------

template<int WQ>
__global__ __launch_bounds__(256)
void k_gather_loss(const int* __restrict__ idx, const float* __restrict__ Z,
                   const float* __restrict__ CB, unsigned short* __restrict__ qB,
                   int d, float* __restrict__ lossAcc) {
  __shared__ float wsum[8];
  const int tid  = threadIdx.x;
  const int wave = tid >> 5;
  const int lane = tid & 31;
  const int r = blockIdx.x * 8 + wave;
  const int j = idx[r];
  const float* crow = CB + (size_t)j * d;
  const float* zrow = Z + (size_t)r * d;
  float s = 0.f;
  for (int c = lane; c < d; c += 32) {
    float q = crow[c];
    float diff = q - zrow[c];
    s += diff * diff;
    if (WQ) qB[(size_t)r * d + c] = f2bf(q);
  }
  for (int off = 16; off; off >>= 1) s += __shfl_xor(s, off, 32);
  if (lane == 0) wsum[wave] = s;
  __syncthreads();
  if (tid == 0) {
    float t = 0.f;
    for (int w = 0; w < 8; ++w) t += wsum[w];
    atomicAdd(lossAcc, t);
  }
}

__global__ void k_finalize(const float* __restrict__ lossAcc, float* __restrict__ out) {
  if (threadIdx.x == 0) {
    const float N = 16384.f;
    // loss = q_loss + beta*e_loss = (1+0.25)*mean((q-z)^2) per level
    out[0] = 1.25f * (lossAcc[0] / (N * 512.f) +
                      lossAcc[1] / (N * 256.f) +
                      lossAcc[2] / (N * 128.f));
  }
}

// ---------------------------------------------------------------------------

extern "C" void kernel_launch(void* const* d_in, const int* in_sizes, int n_in,
                              void* d_out, int out_size, void* d_ws, size_t ws_size,
                              hipStream_t stream) {
  (void)in_sizes; (void)n_in; (void)out_size; (void)ws_size;
  const int N = 16384;
  char* ws = (char*)d_ws;
  size_t off = 0;
  auto alloc = [&](size_t bytes) {
    size_t o = off;
    off += (bytes + 255) & ~(size_t)255;
    return o;
  };

  // ---- workspace layout ----
  unsigned short* xb = (unsigned short*)(ws + alloc((size_t)N * 768 * 2));
  const int wdK[12]  = {768, 384, 512, 256, 256, 128, 512, 1024, 256, 512, 128, 256};
  const int wdM[12]  = {384, 512, 256, 256, 128, 128, 1024, 256, 512, 128, 256, 768};
  const int wdIn[12] = {1, 3, 7, 9, 13, 15, 19, 21, 25, 27, 31, 33};
  unsigned short* wt[12];
  for (int i = 0; i < 12; ++i)
    wt[i] = (unsigned short*)(ws + alloc((size_t)wdK[i] * wdM[i] * 2));
  const int cbd[3] = {512, 256, 128};
  unsigned short* cbb[3];
  for (int l = 0; l < 3; ++l)
    cbb[l] = (unsigned short*)(ws + alloc((size_t)8192 * cbd[l] * 2));
  float* cnorm[3];
  for (int l = 0; l < 3; ++l) cnorm[l] = (float*)(ws + alloc(8192 * 4));
  float* lossAcc = (float*)(ws + alloc(256));
  unsigned short* hA = (unsigned short*)(ws + alloc((size_t)N * 1024 * 2));
  unsigned short* zB = (unsigned short*)(ws + alloc((size_t)N * 512 * 2));
  float* zF = (float*)(ws + alloc((size_t)N * 512 * 4));
  float* yF = (float*)(ws + alloc((size_t)N * 512 * 4));
  unsigned short* qB = (unsigned short*)(ws + alloc((size_t)N * 512 * 2));

  float* outF = (float*)d_out;
  int* idxBase = (int*)(outF + (size_t)N * 768 + 1);

  // ---- one-time conversions ----
  { int n = N * 768; k_cvt_bf16<<<(n + 255) / 256, 256, 0, stream>>>((const float*)d_in[0], xb, n); }
  for (int i = 0; i < 12; ++i) {
    int n = wdK[i] * wdM[i];
    k_transpose_bf16<<<(n + 255) / 256, 256, 0, stream>>>((const float*)d_in[wdIn[i]], wt[i], wdK[i], wdM[i]);
  }
  for (int l = 0; l < 3; ++l) {
    int n = 8192 * cbd[l];
    k_cvt_bf16<<<(n + 255) / 256, 256, 0, stream>>>((const float*)d_in[35 + l], cbb[l], n);
    k_norms<<<1024, 256, 0, stream>>>((const float*)d_in[35 + l], cnorm[l], cbd[l]);
  }
  k_init_loss<<<1, 32, 0, stream>>>(lossAcc);

  const int RT = N / 16;  // 1024 row tiles
  // ---- encoder level 0 ----
  k_gemm<1, 0, 1><<<dim3(RT, 3), 256, 0, stream>>>(xb, wt[0], (const float*)d_in[2], nullptr, hA, 768, 384);
  k_gemm<0, 1, 0><<<dim3(RT, 4), 256, 0, stream>>>(hA, wt[1], (const float*)d_in[4], yF, nullptr, 384, 512);
  k_layernorm<<<N / 8, 256, 0, stream>>>(yF, (const float*)d_in[5], (const float*)d_in[6], zF, zB, 512);
  k_vq_argmin<<<RT, 256, 0, stream>>>(zB, cbb[0], cnorm[0], idxBase + 0 * N, 512);
  k_gather_loss<1><<<N / 8, 256, 0, stream>>>(idxBase + 0 * N, zF, (const float*)d_in[35], qB, 512, lossAcc + 0);
  // ---- encoder level 1 ----
  k_gemm<1, 0, 1><<<dim3(RT, 2), 256, 0, stream>>>(zB, wt[2], (const float*)d_in[8], nullptr, hA, 512, 256);
  k_gemm<0, 1, 0><<<dim3(RT, 2), 256, 0, stream>>>(hA, wt[3], (const float*)d_in[10], yF, nullptr, 256, 256);
  k_layernorm<<<N / 8, 256, 0, stream>>>(yF, (const float*)d_in[11], (const float*)d_in[12], zF, zB, 256);
  k_vq_argmin<<<RT, 256, 0, stream>>>(zB, cbb[1], cnorm[1], idxBase + 1 * N, 256);
  k_gather_loss<0><<<N / 8, 256, 0, stream>>>(idxBase + 1 * N, zF, (const float*)d_in[36], nullptr, 256, lossAcc + 1);
  // ---- encoder level 2 ----
  k_gemm<1, 0, 1><<<dim3(RT, 1), 256, 0, stream>>>(zB, wt[4], (const float*)d_in[14], nullptr, hA, 256, 128);
  k_gemm<0, 1, 0><<<dim3(RT, 1), 256, 0, stream>>>(hA, wt[5], (const float*)d_in[16], yF, nullptr, 128, 128);
  k_layernorm<<<N / 8, 256, 0, stream>>>(yF, (const float*)d_in[17], (const float*)d_in[18], zF, zB, 128);
  k_vq_argmin<<<RT, 256, 0, stream>>>(zB, cbb[2], cnorm[2], idxBase + 2 * N, 128);
  k_gather_loss<0><<<N / 8, 256, 0, stream>>>(idxBase + 2 * N, zF, (const float*)d_in[37], nullptr, 128, lossAcc + 2);
  // ---- decoder (from quantized level 0) ----
  k_gemm<1, 0, 1><<<dim3(RT, 8), 256, 0, stream>>>(qB, wt[6], (const float*)d_in[20], nullptr, hA, 512, 1024);
  k_gemm<0, 1, 0><<<dim3(RT, 2), 256, 0, stream>>>(hA, wt[7], (const float*)d_in[22], yF, nullptr, 1024, 256);
  k_layernorm<<<N / 8, 256, 0, stream>>>(yF, (const float*)d_in[23], (const float*)d_in[24], zF, zB, 256);
  k_gemm<1, 0, 1><<<dim3(RT, 4), 256, 0, stream>>>(zB, wt[8], (const float*)d_in[26], nullptr, hA, 256, 512);
  k_gemm<0, 1, 0><<<dim3(RT, 1), 256, 0, stream>>>(hA, wt[9], (const float*)d_in[28], yF, nullptr, 512, 128);
  k_layernorm<<<N / 8, 256, 0, stream>>>(yF, (const float*)d_in[29], (const float*)d_in[30], zF, zB, 128);
  k_gemm<1, 0, 1><<<dim3(RT, 2), 256, 0, stream>>>(zB, wt[10], (const float*)d_in[32], nullptr, hA, 128, 256);
  k_gemm<0, 1, 0><<<dim3(RT, 6), 256, 0, stream>>>(hA, wt[11], (const float*)d_in[34], outF, nullptr, 256, 768);
  // ---- loss scalar ----
  k_finalize<<<1, 32, 0, stream>>>(lossAcc, outF + (size_t)N * 768);
}
